// MultiHead_59193239274209
// MI455X (gfx1250) — compile-verified, matrix operand order checked by reading, not run on hardware
//
#include <hip/hip_runtime.h>
#include <cstdint>

typedef __attribute__((ext_vector_type(16))) _Float16 v16h;
typedef __attribute__((ext_vector_type(8)))  float    v8f;

constexpr int   BB = 64, T = 1024, D = 128;
constexpr int   QB = 32;                          // queries per block (2 subtiles of 16)
constexpr float MASK_FILL = -4294967295.0f;       // float(-2^32 + 1)
constexpr float SCALE     = 0.08838834764831845f; // 1/sqrt(128)

constexpr int SSTR = T + 8;      // score row stride (break 64-bank alignment)
constexpr int OSTR = D + 8;      // output tile row stride
constexpr int SMEM_FLOATS = QB * SSTR + QB * OSTR + QB * 8 + QB + QB;

// f16 A/B fragment element -> K index (ISA 16-bit 16x32 layout):
// lanes 0-15 hold K {0..7,16..23}, lanes 16-31 hold K {8..15,24..31}
__device__ __forceinline__ int kdmap(int e, int kh) {
    return e + 8 * kh + ((e >= 8) ? 8 : 0);
}

__global__ __launch_bounds__(256)
void mha_fwd_kernel(const float* __restrict__ key,
                    const float* __restrict__ value,
                    const float* __restrict__ query,
                    const unsigned char* __restrict__ mask,
                    const float* __restrict__ qmask,
                    float* __restrict__ out)
{
    extern __shared__ float smem[];
    float* sS   = smem;                  // QB x SSTR : score/prob strip
    float* sO   = sS + QB * SSTR;        // QB x OSTR : output accumulator
    float* sRed = sO + QB * OSTR;        // QB x 8    : row-reduction scratch
    float* sMax = sRed + QB * 8;         // QB
    float* sSum = sMax + QB;             // QB

    const int tid  = threadIdx.x;
    const int lane = tid & 31;
    const int wave = tid >> 5;           // 8 waves, each owns 128 key columns
    const int m    = lane & 15;          // A-row / B-col index inside fragment
    const int kh   = lane >> 4;          // K-half selector
    const int wb   = wave * 128;         // this wave's key-column window

    const int b  = (int)blockIdx.x >> 5;         // 32 blocks per batch
    const int q0 = ((int)blockIdx.x & 31) << 5;  // 32-query tile base

    const float*   Kp = key   + (size_t)b * T * D;
    const float*   Vp = value + (size_t)b * T * D;
    const float*   Qp = query + ((size_t)b * T + q0) * D;
    const uint8_t* Mp = mask  + ((size_t)b * T + q0) * T;
    const float*   qm = qmask + (size_t)b * T + q0;
    float* resOut  = out + ((size_t)b * T + q0) * D;
    float* attnOut = out + (size_t)BB * T * D + ((size_t)b * T + q0) * T;

    // zero the cross-wave output accumulator (consumed after later barriers)
    for (int i = tid; i < QB * OSTR; i += 256) sO[i] = 0.f;

    // warm L2/L0 with this wave's V block while we do the score phase
    #pragma unroll
    for (int pf = 0; pf < 4; ++pf)
        __builtin_prefetch(&Vp[(size_t)(wb + pf * 32 + lane) * D], 0, 1);

    // ---------------- Phase A: S = scale * Q K^T -> LDS (unmasked) ----------------
    v16h aQ[2][4];                        // two 16x128 query subtiles, 4 K-chunks each
    #pragma unroll
    for (int s = 0; s < 2; ++s)
        #pragma unroll
        for (int c = 0; c < 4; ++c)
            #pragma unroll
            for (int e = 0; e < 16; ++e)
                aQ[s][c][e] = (_Float16)(Qp[(s * 16 + m) * D + 32 * c + kdmap(e, kh)] * SCALE);

    for (int j = 0; j < 8; ++j) {
        const int n0 = (wave * 8 + j) * 16;          // key-tile base column
        v8f a0 = {}, a1 = {};
        #pragma unroll
        for (int cc = 0; cc < 4; ++cc) {
            v16h bK;                                  // B = K^T chunk: lane = col n, elems = d
            #pragma unroll
            for (int e = 0; e < 16; ++e)
                bK[e] = (_Float16)Kp[(size_t)(n0 + m) * D + 32 * cc + kdmap(e, kh)];
            a0 = __builtin_amdgcn_wmma_f32_16x16x32_f16(false, aQ[0][cc], false, bK,
                                                        (short)0, a0, false, false);
            a1 = __builtin_amdgcn_wmma_f32_16x16x32_f16(false, aQ[1][cc], false, bK,
                                                        (short)0, a1, false, false);
        }
        // C/D layout: vgpr r -> row r (+8 for upper lane half); col = lane%16
        #pragma unroll
        for (int r = 0; r < 8; ++r) {
            const int col = n0 + m;
            sS[(r + 8 * kh) * SSTR + col]      = a0[r];
            sS[(16 + r + 8 * kh) * SSTR + col] = a1[r];
        }
    }
    __syncthreads();

    // ------------- Phase B: apply mask (vector loads) + softmax statistics -------------
    const int rrow = tid >> 3;            // 32 rows x 8 threads each
    const int sub  = tid & 7;
    float mx = -3.0e38f;
    for (int c = 0; c < 8; ++c) {
        const int col0 = (sub + 8 * c) * 16;          // 16-byte-aligned mask chunk
        uint8_t mloc[16];
        *(uint4*)mloc = *(const uint4*)&Mp[(size_t)rrow * T + col0];  // B128 mask load
        #pragma unroll
        for (int i = 0; i < 16; ++i) {
            float s = sS[rrow * SSTR + col0 + i];
            if (mloc[i]) s = MASK_FILL;
            sS[rrow * SSTR + col0 + i] = s;
            mx = fmaxf(mx, s);
        }
    }
    sRed[rrow * 8 + sub] = mx;
    __syncthreads();
    if (sub == 0) {
        float v = sRed[rrow * 8];
        #pragma unroll
        for (int i = 1; i < 8; ++i) v = fmaxf(v, sRed[rrow * 8 + i]);
        sMax[rrow] = v;
    }
    __syncthreads();
    const float rowmax = sMax[rrow];
    float sum = 0.f;
    for (int c = 0; c < 8; ++c) {
        const int col0 = (sub + 8 * c) * 16;
        #pragma unroll
        for (int i = 0; i < 16; ++i) {
            float p = __expf(sS[rrow * SSTR + col0 + i] - rowmax);
            sS[rrow * SSTR + col0 + i] = p;           // strip now holds unnormalized probs
            sum += p;
        }
    }
    sRed[rrow * 8 + sub] = sum;
    __syncthreads();
    if (sub == 0) {
        float v = 0.f;
        #pragma unroll
        for (int i = 0; i < 8; ++i) v += sRed[rrow * 8 + i];
        sSum[rrow] = v;
    }
    __syncthreads();

    // ---------------- Phase C1: attn out = p/sum * query_mask ----------------
    for (int r = 0; r < QB; ++r) {
        const float qs = qm[r] / sSum[r];
        const int col = wb + lane * 4;
        float4 p4 = *(const float4*)&sS[r * SSTR + col];
        p4.x *= qs; p4.y *= qs; p4.z *= qs; p4.w *= qs;
        *(float4*)&attnOut[(size_t)r * T + col] = p4; // 128 contiguous floats per wave row
    }

    // ---------------- Phase C2: O += P V over this wave's 128 keys ----------------
    const float qs0 = qm[m]      / sSum[m];
    const float qs1 = qm[16 + m] / sSum[16 + m];
    v16h aP[2][4];
    #pragma unroll
    for (int j2 = 0; j2 < 4; ++j2) {
        #pragma unroll
        for (int e = 0; e < 16; ++e) {
            const int col = wb + 32 * j2 + kdmap(e, kh);
            aP[0][j2][e] = (_Float16)(sS[m * SSTR + col]        * qs0);
            aP[1][j2][e] = (_Float16)(sS[(16 + m) * SSTR + col] * qs1);
        }
    }

    for (int dt = 0; dt < 8; ++dt) {      // 8 output d-tiles of 16
        v8f a0 = {}, a1 = {};
        #pragma unroll
        for (int j2 = 0; j2 < 4; ++j2) {
            v16h bV;                       // B = V chunk: lane = d col, elems = k
            #pragma unroll
            for (int e = 0; e < 16; ++e)
                bV[e] = (_Float16)Vp[(size_t)(wb + 32 * j2 + kdmap(e, kh)) * D + 16 * dt + m];
            a0 = __builtin_amdgcn_wmma_f32_16x16x32_f16(false, aP[0][j2], false, bV,
                                                        (short)0, a0, false, false);
            a1 = __builtin_amdgcn_wmma_f32_16x16x32_f16(false, aP[1][j2], false, bV,
                                                        (short)0, a1, false, false);
        }
        #pragma unroll
        for (int r = 0; r < 8; ++r) {
            atomicAdd(&sO[(r + 8 * kh) * OSTR + 16 * dt + m],        a0[r]);  // ds_add_f32
            atomicAdd(&sO[(16 + r + 8 * kh) * OSTR + 16 * dt + m],   a1[r]);
        }
    }
    __syncthreads();

    // ---------------- write result tile (32 x 128) ----------------
    {
        const int row = tid >> 3;
        const int d0  = (tid & 7) * 16;
        #pragma unroll
        for (int v = 0; v < 4; ++v) {
            float4 x = *(const float4*)&sO[row * OSTR + d0 + 4 * v];
            *(float4*)&resOut[(size_t)row * D + d0 + 4 * v] = x;
        }
    }
}

extern "C" void kernel_launch(void* const* d_in, const int* in_sizes, int n_in,
                              void* d_out, int out_size, void* d_ws, size_t ws_size,
                              hipStream_t stream) {
    (void)in_sizes; (void)n_in; (void)out_size; (void)d_ws; (void)ws_size;
    const float*         key   = (const float*)d_in[0];
    const float*         value = (const float*)d_in[1];
    const float*         query = (const float*)d_in[2];
    const unsigned char* mask  = (const unsigned char*)d_in[3];  // jnp.bool_: 1 byte/elem
    const float*         qmask = (const float*)d_in[4];
    float* out = (float*)d_out;   // [result (B*T*D) | attn (B*T*T)] concatenated

    const size_t smem = (size_t)SMEM_FLOATS * sizeof(float);   // ~147 KB dynamic LDS
    hipFuncSetAttribute((const void*)mha_fwd_kernel,
                        hipFuncAttributeMaxDynamicSharedMemorySize, (int)smem);
    // one block per (batch, 32-query tile): 64 * 32 = 2048 blocks, 8 wave32 each
    mha_fwd_kernel<<<BB * (T / QB), 256, smem, stream>>>(key, value, query, mask, qmask, out);
}
